// TinyFluxDeep_82686710383292
// MI455X (gfx1250) — matrix operand order repaired and checked by reading, not compiled, not run on hardware
//
// TinyFluxDeep dual-stream DiT block for MI455X (gfx1250, wave32, WMMA bf16,
// async global->LDS double-buffered GEMM).
#include <hip/hip_runtime.h>
#include <cstdint>
#include <cstddef>

#define CB   2
#define CL   512
#define CN   2048
#define CT   2560      // CL + CN
#define CD   512
#define CNH  4
#define CHD  128
#define CFF  2048
#define CSCALE 0.088388347648318447f   // 128^-0.5

typedef __attribute__((ext_vector_type(16))) __bf16 v16bf;
typedef __attribute__((ext_vector_type(8)))  float  v8f;

union Frag { uint4 q[2]; v16bf v; };

__device__ __forceinline__ unsigned short f2bf(float f) {
  unsigned int u = __float_as_uint(f);
  u += 0x7fffu + ((u >> 16) & 1u);          // round-to-nearest-even
  return (unsigned short)(u >> 16);
}

__device__ __forceinline__ v8f wmma_bf(const Frag& a, const Frag& b, v8f c) {
  return __builtin_amdgcn_wmma_f32_16x16x32_bf16(false, a.v, false, b.v,
                                                 (short)0, c, false, false);
}

__device__ __forceinline__ float gelu_tanh(float x) {
  float x3 = x * x * x;
  return 0.5f * x * (1.f + tanhf(0.79788456080286535f * (x + 0.044715f * x3)));
}

// Async DMA: copy 32 bytes (two b128) global -> LDS, tracked by ASYNCcnt.
// INST_OFFSET applies to both LDS and memory addresses (ISA 08_async_tensor §4.4).
__device__ __forceinline__ void async_copy32(unsigned int lds_addr,
                                             const void* gptr) {
  unsigned long long ga = (unsigned long long)gptr;
  asm volatile("global_load_async_to_lds_b128 %0, %1, off\n\t"
               "global_load_async_to_lds_b128 %0, %1, off offset:16"
               :: "v"(lds_addr), "v"(ga) : "memory");
}
__device__ __forceinline__ void wait_async0() {
  asm volatile("s_wait_asynccnt 0x0" ::: "memory");
}

// ---------------------------------------------------------------- conversions
__global__ __launch_bounds__(256) void f32_to_bf16_kernel(
    const float* __restrict__ in, unsigned short* __restrict__ out, int n) {
  int i = blockIdx.x * 256 + threadIdx.x;
  if (i < n) out[i] = f2bf(in[i]);
}

// ---------------------------------------------------- adaLN embedding (small)
__global__ __launch_bounds__(256) void adaln_emb_kernel(
    const float* __restrict__ vec, const float* __restrict__ Wm,
    const float* __restrict__ bias, float* __restrict__ e) {
  int o = blockIdx.x * 8 + (threadIdx.x >> 5);      // one wave per output
  int lane = threadIdx.x & 31;
  int b = o / (6 * CD), j = o % (6 * CD);
  const float* v = vec + (size_t)b * CD;
  const float* w = Wm + (size_t)j * CD;
  float s = 0.f;
  for (int d = lane; d < CD; d += 32) {
    float x = v[d];
    s += (x / (1.f + __expf(-x))) * w[d];
  }
  #pragma unroll
  for (int off = 16; off > 0; off >>= 1) s += __shfl_xor(s, off, 32);
  if (lane == 0) e[o] = s + bias[j];
}

// ------------------------------------------------------- temperature -> eff_b
__global__ void eff_kernel(const float* __restrict__ temp, float* __restrict__ eff) {
  int b = threadIdx.x;
  if (b < CB) {
    float m = 0.f;
    for (int h = 0; h < CNH; ++h) m += temp[b * CNH + h];
    m = fmaxf(m * (1.f / CNH), 0.1f);
    eff[b] = CSCALE / m;
  }
}

// ------------------------------------------ spatial modulation (bilinear 8x8)
__global__ __launch_bounds__(256) void mod_kernel(
    const float* __restrict__ sp, const float* __restrict__ mw,
    const float* __restrict__ mb, float* __restrict__ mod) {
  int idx = blockIdx.x * 256 + threadIdx.x;          // B*NH*N = 16384
  int n = idx % CN;
  int h = (idx / CN) % CNH;
  int b = idx / (CN * CNH);
  int i = n >> 5, j = n & 31;                        // H=64, W=32
  float y = (i + 0.5f) * 0.125f - 0.5f;
  float x = (j + 0.5f) * 0.25f  - 0.5f;
  int y0 = (int)floorf(y), x0 = (int)floorf(x);
  float fy = y - y0, fx = x - x0;
  int y0c = min(max(y0, 0), 7), y1c = min(max(y0 + 1, 0), 7);
  int x0c = min(max(x0, 0), 7), x1c = min(max(x0 + 1, 0), 7);
  const float* s = sp + (size_t)b * 64;
  float v00 = s[y0c * 8 + x0c], v01 = s[y0c * 8 + x1c];
  float v10 = s[y1c * 8 + x0c], v11 = s[y1c * 8 + x1c];
  float v = (1.f - fy) * ((1.f - fx) * v00 + fx * v01) +
            fy * ((1.f - fx) * v10 + fx * v11);
  float m = v * mw[h] + mb[h];
  m = fminf(fmaxf(m, -2.f), 2.f);
  mod[idx] = __expf(m);
}

// ----------------------------------- RMSNorm + adaLN shift/scale -> bf16 rows
__global__ __launch_bounds__(256) void norm_mod_kernel(
    const float* __restrict__ X, const float* __restrict__ e,
    const float* __restrict__ nw, unsigned short* __restrict__ Out,
    int rows_per_batch, int shift_off, int scale_off) {
  int row = blockIdx.x;
  int b = row / rows_per_batch;
  const float* x = X + (size_t)row * CD;
  int t = threadIdx.x;
  float v0 = x[t], v1 = x[t + 256];
  float s = v0 * v0 + v1 * v1;
  #pragma unroll
  for (int off = 16; off > 0; off >>= 1) s += __shfl_xor(s, off, 32);
  __shared__ float red[8];
  int wave = t >> 5, lane = t & 31;
  if (lane == 0) red[wave] = s;
  __syncthreads();
  float tot = red[0] + red[1] + red[2] + red[3] + red[4] + red[5] + red[6] + red[7];
  float inv = rsqrtf(tot * (1.f / CD) + 1e-6f);
  const float* eb = e + (size_t)b * 6 * CD;
  unsigned short* o = Out + (size_t)row * CD;
  float y0 = v0 * inv * nw[t]       * (1.f + eb[scale_off + t])       + eb[shift_off + t];
  float y1 = v1 * inv * nw[t + 256] * (1.f + eb[scale_off + t + 256]) + eb[shift_off + t + 256];
  o[t] = f2bf(y0);
  o[t + 256] = f2bf(y1);
}

// ------------------------------------------------------------ WMMA GEMM ------
// C[M,N] = A_bf16[M,K] * W_bf16[N,K]^T ; block tile 128x128, 8 waves, each wave
// 32x64 (2x4 WMMA 16x16). Double-buffered async global->LDS tile copies.
// epi: 0 = store f32; 1 = gelu(acc+bias) -> bf16; 2 = res + gate*(acc[+bias]) -> f32
#define LDT 40   // padded LDS row stride (bf16 units)
__global__ __launch_bounds__(256) void gemm_bf16_kernel(
    const unsigned short* __restrict__ A, const unsigned short* __restrict__ W,
    const float* __restrict__ bias, float* __restrict__ Cf,
    unsigned short* __restrict__ Cb, const float* __restrict__ res,
    const float* __restrict__ gate, int M, int Ncols, int Kdim, int epi,
    int rows_per_batch) {
  __shared__ unsigned short As[2][128 * LDT];
  __shared__ unsigned short Ws[2][128 * LDT];
  int t = threadIdx.x;
  int lane = t & 31, wave = t >> 5;
  int wm = wave >> 1, wn = wave & 1;
  int mlo = lane & 15, half = lane >> 4;
  int blockM = blockIdx.y * 128, blockN = blockIdx.x * 128;

  v8f acc[2][4];
  #pragma unroll
  for (int i = 0; i < 2; ++i)
    #pragma unroll
    for (int j = 0; j < 4; ++j)
      acc[i][j] = (v8f){0.f, 0.f, 0.f, 0.f, 0.f, 0.f, 0.f, 0.f};

  int ldRow = t >> 1, ldCol = (t & 1) * 16;
  const unsigned short* gA0 = A + (size_t)(blockM + ldRow) * Kdim + ldCol;
  const unsigned short* gW0 = W + (size_t)(blockN + ldRow) * Kdim + ldCol;

  auto issue_tile = [&](int k0, int buf) {
    async_copy32((unsigned int)(uintptr_t)&As[buf][ldRow * LDT + ldCol], gA0 + k0);
    async_copy32((unsigned int)(uintptr_t)&Ws[buf][ldRow * LDT + ldCol], gW0 + k0);
  };

  issue_tile(0, 0);
  int cur = 0;
  for (int k0 = 0; k0 < Kdim; k0 += 32) {
    wait_async0();          // own async writes (tile 'cur') complete
    __syncthreads();        // all waves' writes complete -> safe to read 'cur'
    if (k0 + 32 < Kdim) issue_tile(k0 + 32, cur ^ 1);

    Frag af[2], wf[4];
    #pragma unroll
    for (int i = 0; i < 2; ++i) {
      const unsigned short* p = &As[cur][(wm * 32 + i * 16 + mlo) * LDT + half * 8];
      af[i].q[0] = *(const uint4*)p;
      af[i].q[1] = *(const uint4*)(p + 16);
    }
    #pragma unroll
    for (int j = 0; j < 4; ++j) {
      const unsigned short* p = &Ws[cur][(wn * 64 + j * 16 + mlo) * LDT + half * 8];
      wf[j].q[0] = *(const uint4*)p;
      wf[j].q[1] = *(const uint4*)(p + 16);
    }
    #pragma unroll
    for (int i = 0; i < 2; ++i)
      #pragma unroll
      for (int j = 0; j < 4; ++j)
        acc[i][j] = wmma_bf(af[i], wf[j], acc[i][j]);
    cur ^= 1;
  }

  #pragma unroll
  for (int i = 0; i < 2; ++i)
    #pragma unroll
    for (int j = 0; j < 4; ++j)
      #pragma unroll
      for (int r = 0; r < 8; ++r) {
        int row = blockM + wm * 32 + i * 16 + 8 * half + r;
        int col = blockN + wn * 64 + j * 16 + mlo;
        float v = acc[i][j][r];
        if (epi == 0) {
          Cf[(size_t)row * Ncols + col] = v;
        } else if (epi == 1) {
          Cb[(size_t)row * Ncols + col] = f2bf(gelu_tanh(v + bias[col]));
        } else {
          if (bias) v += bias[col];
          int bb = row / rows_per_batch;
          Cf[(size_t)row * Ncols + col] =
              res[(size_t)row * Ncols + col] + gate[(size_t)bb * 6 * CD + col] * v;
        }
      }
}

// ----------------------------- assemble Q/K/V (rope + spatial mod, bf16, V^T)
__global__ __launch_bounds__(128) void assemble_kernel(
    const float* __restrict__ tqkv, const float* __restrict__ iqkv,
    const float* __restrict__ rope, const float* __restrict__ mod,
    unsigned short* __restrict__ Q, unsigned short* __restrict__ K,
    unsigned short* __restrict__ Vt) {
  int d = threadIdx.x;                 // 0..127
  int tk = blockIdx.x;                 // 0..T-1
  int h = blockIdx.y, b = blockIdx.z;
  float q, k, v;
  if (tk < CL) {
    const float* rowp = tqkv + (size_t)(b * CL + tk) * (3 * CNH * CHD);
    q = rowp[(0 * CNH + h) * CHD + d];
    k = rowp[(1 * CNH + h) * CHD + d];
    v = rowp[(2 * CNH + h) * CHD + d];
  } else {
    int n = tk - CL;
    const float* rowp = iqkv + (size_t)(b * CN + n) * (3 * CNH * CHD);
    float q0 = rowp[(0 * CNH + h) * CHD + d];
    float q1 = rowp[(0 * CNH + h) * CHD + (d ^ 1)];
    float k0 = rowp[(1 * CNH + h) * CHD + d];
    float k1 = rowp[(1 * CNH + h) * CHD + (d ^ 1)];
    v = rowp[(2 * CNH + h) * CHD + d];
    float c = rope[(size_t)n * CHD + (d & ~1)];
    float s = rope[(size_t)n * CHD + (d & ~1) + 1];
    float m = mod[(size_t)(b * CNH + h) * CN + n];
    if ((d & 1) == 0) { q = q0 * c - q1 * s; k = k0 * c - k1 * s; }
    else              { q = q0 * c + q1 * s; k = k0 * c + k1 * s; }
    q *= m; k *= m;
  }
  size_t qk = ((size_t)(b * CNH + h) * CT + tk) * CHD + d;
  Q[qk] = f2bf(q);
  K[qk] = f2bf(k);
  Vt[((size_t)(b * CNH + h) * CHD + d) * CT + tk] = f2bf(v);
}

// -------------------------------------------------- flash attention (1 wave)
__global__ __launch_bounds__(32) void attn_kernel(
    const unsigned short* __restrict__ Q, const unsigned short* __restrict__ K,
    const unsigned short* __restrict__ Vt, const float* __restrict__ eff,
    unsigned short* __restrict__ txt_attn, unsigned short* __restrict__ img_attn) {
  __shared__ unsigned short Ps[16 * LDT];
  int lane = threadIdx.x;
  int q0 = blockIdx.x * 16;
  int h = blockIdx.y, b = blockIdx.z;
  int mlo = lane & 15, half = lane >> 4;
  const unsigned short* Qb = Q + ((size_t)(b * CNH + h) * CT + q0) * CHD;
  const unsigned short* Kb = K + (size_t)(b * CNH + h) * CT * CHD;
  const unsigned short* Vb = Vt + (size_t)(b * CNH + h) * CHD * CT;
  float sc = (q0 < CL) ? CSCALE : eff[b];

  Frag qf[4];
  #pragma unroll
  for (int ks = 0; ks < 4; ++ks) {
    const unsigned short* p = Qb + (size_t)mlo * CHD + ks * 32 + half * 8;
    qf[ks].q[0] = *(const uint4*)p;
    qf[ks].q[1] = *(const uint4*)(p + 16);
  }

  v8f o[8];
  float mR[8], lR[8];
  #pragma unroll
  for (int i = 0; i < 8; ++i) {
    o[i] = (v8f){0.f, 0.f, 0.f, 0.f, 0.f, 0.f, 0.f, 0.f};
    mR[i] = -1e30f;
    lR[i] = 0.f;
  }

  for (int key0 = 0; key0 < CT; key0 += 32) {
    // prefetch next key/value tiles (global_prefetch_b8)
    if (key0 + 32 < CT) {
      __builtin_prefetch(Kb + (size_t)(key0 + 32 + mlo) * CHD, 0, 0);
      __builtin_prefetch(Vb + (size_t)(lane * 4) * CT + key0 + 32, 0, 0);
    }
    v8f s0 = (v8f){0.f, 0.f, 0.f, 0.f, 0.f, 0.f, 0.f, 0.f};
    v8f s1 = (v8f){0.f, 0.f, 0.f, 0.f, 0.f, 0.f, 0.f, 0.f};
    #pragma unroll
    for (int ks = 0; ks < 4; ++ks) {
      Frag kf0, kf1;
      const unsigned short* p0 = Kb + (size_t)(key0 + mlo) * CHD + ks * 32 + half * 8;
      kf0.q[0] = *(const uint4*)p0;
      kf0.q[1] = *(const uint4*)(p0 + 16);
      const unsigned short* p1 = p0 + 16 * CHD;
      kf1.q[0] = *(const uint4*)p1;
      kf1.q[1] = *(const uint4*)(p1 + 16);
      s0 = wmma_bf(qf[ks], kf0, s0);
      s1 = wmma_bf(qf[ks], kf1, s1);
    }
    float al[8];
    #pragma unroll
    for (int r = 0; r < 8; ++r) {
      float a = s0[r] * sc, bv = s1[r] * sc;
      float v = fmaxf(a, bv);
      #pragma unroll
      for (int off = 1; off < 16; off <<= 1) v = fmaxf(v, __shfl_xor(v, off, 32));
      float nm = fmaxf(mR[r], v);
      al[r] = __expf(mR[r] - nm);
      mR[r] = nm;
      float p0v = __expf(a - nm), p1v = __expf(bv - nm);
      float rs = p0v + p1v;
      #pragma unroll
      for (int off = 1; off < 16; off <<= 1) rs += __shfl_xor(rs, off, 32);
      lR[r] = lR[r] * al[r] + rs;
      Ps[(r + 8 * half) * LDT + mlo] = f2bf(p0v);
      Ps[(r + 8 * half) * LDT + 16 + mlo] = f2bf(p1v);
    }
    #pragma unroll
    for (int nt = 0; nt < 8; ++nt)
      #pragma unroll
      for (int r = 0; r < 8; ++r)
        o[nt][r] *= al[r];
    Frag pa;
    const unsigned short* pp = &Ps[mlo * LDT + half * 8];
    pa.q[0] = *(const uint4*)pp;
    pa.q[1] = *(const uint4*)(pp + 16);
    #pragma unroll
    for (int nt = 0; nt < 8; ++nt) {
      Frag vf;
      const unsigned short* vp = Vb + (size_t)(nt * 16 + mlo) * CT + key0 + half * 8;
      vf.q[0] = *(const uint4*)vp;
      vf.q[1] = *(const uint4*)(vp + 16);
      o[nt] = wmma_bf(pa, vf, o[nt]);
    }
  }

  #pragma unroll
  for (int nt = 0; nt < 8; ++nt)
    #pragma unroll
    for (int r = 0; r < 8; ++r) {
      float v = o[nt][r] / lR[r];
      int m = r + 8 * half;
      int col = h * CHD + nt * 16 + mlo;
      unsigned short bv = f2bf(v);
      if (q0 < CL)
        txt_attn[(size_t)(b * CL + q0 + m) * (CNH * CHD) + col] = bv;
      else
        img_attn[(size_t)(b * CN + (q0 - CL) + m) * (CNH * CHD) + col] = bv;
    }
}

// ============================================================================
extern "C" void kernel_launch(void* const* d_in, const int* in_sizes, int n_in,
                              void* d_out, int out_size, void* d_ws, size_t ws_size,
                              hipStream_t stream) {
  const float* txt   = (const float*)d_in[0];
  const float* img   = (const float*)d_in[1];
  const float* vec   = (const float*)d_in[2];
  const float* rope  = (const float*)d_in[3];
  const float* soltp = (const float*)d_in[4];
  const float* solsp = (const float*)d_in[5];
  const float* i_aw  = (const float*)d_in[6];
  const float* i_ab  = (const float*)d_in[7];
  const float* i_anw = (const float*)d_in[8];
  const float* t_aw  = (const float*)d_in[9];
  const float* t_ab  = (const float*)d_in[10];
  const float* t_anw = (const float*)d_in[11];
  const float* t_qkv = (const float*)d_in[12];
  const float* i_qkv = (const float*)d_in[13];
  const float* t_ow  = (const float*)d_in[14];
  const float* i_ow  = (const float*)d_in[15];
  const float* modw  = (const float*)d_in[16];
  const float* modb  = (const float*)d_in[17];
  const float* i_n2w = (const float*)d_in[18];
  const float* t_n2w = (const float*)d_in[19];
  const float* i_f1w = (const float*)d_in[20];
  const float* i_f1b = (const float*)d_in[21];
  const float* i_f2w = (const float*)d_in[22];
  const float* i_f2b = (const float*)d_in[23];
  const float* t_f1w = (const float*)d_in[24];
  const float* t_f1b = (const float*)d_in[25];
  const float* t_f2w = (const float*)d_in[26];
  const float* t_f2b = (const float*)d_in[27];
  float* out = (float*)d_out;

  uint8_t* ws = (uint8_t*)d_ws;
  size_t off = 0;
  auto alloc = [&](size_t bytes) -> void* {
    void* p = ws + off;
    off = (off + bytes + 255) & ~(size_t)255;
    return p;
  };

  float* e_txt = (float*)alloc((size_t)CB * 6 * CD * 4);
  float* e_img = (float*)alloc((size_t)CB * 6 * CD * 4);
  float* effb  = (float*)alloc(256);
  float* modbf = (float*)alloc((size_t)CB * CNH * CN * 4);
  unsigned short* wq_t = (unsigned short*)alloc((size_t)1536 * 512 * 2);
  unsigned short* wq_i = (unsigned short*)alloc((size_t)1536 * 512 * 2);
  unsigned short* wo_t = (unsigned short*)alloc((size_t)512 * 512 * 2);
  unsigned short* wo_i = (unsigned short*)alloc((size_t)512 * 512 * 2);
  unsigned short* w1_t = (unsigned short*)alloc((size_t)CFF * CD * 2);
  unsigned short* w1_i = (unsigned short*)alloc((size_t)CFF * CD * 2);
  unsigned short* w2_t = (unsigned short*)alloc((size_t)CD * CFF * 2);
  unsigned short* w2_i = (unsigned short*)alloc((size_t)CD * CFF * 2);
  unsigned short* txtn = (unsigned short*)alloc((size_t)CB * CL * CD * 2);
  unsigned short* imgn = (unsigned short*)alloc((size_t)CB * CN * CD * 2);
  float* tqkv = (float*)alloc((size_t)CB * CL * 1536 * 4);
  float* iqkv = (float*)alloc((size_t)CB * CN * 1536 * 4);
  unsigned short* Qb  = (unsigned short*)alloc((size_t)CB * CNH * CT * CHD * 2);
  unsigned short* Kb  = (unsigned short*)alloc((size_t)CB * CNH * CT * CHD * 2);
  unsigned short* Vtb = (unsigned short*)alloc((size_t)CB * CNH * CHD * CT * 2);
  unsigned short* tattn = (unsigned short*)alloc((size_t)CB * CL * CD * 2);
  unsigned short* iattn = (unsigned short*)alloc((size_t)CB * CN * CD * 2);
  float* txt1 = (float*)alloc((size_t)CB * CL * CD * 4);
  float* img1 = (float*)alloc((size_t)CB * CN * CD * 4);
  unsigned short* txtin = (unsigned short*)alloc((size_t)CB * CL * CD * 2);
  unsigned short* imgin = (unsigned short*)alloc((size_t)CB * CN * CD * 2);
  // fc1 outputs reuse qkv f32 buffers (qkv consumed before fc1 runs)
  unsigned short* h_t = (unsigned short*)tqkv;   // [CB*CL, FF] bf16 <= tqkv bytes
  unsigned short* h_i = (unsigned short*)iqkv;   // [CB*CN, FF] bf16 <= iqkv bytes

  // 1. weight conversions f32 -> bf16
  auto conv = [&](const float* src, unsigned short* dst, int n) {
    f32_to_bf16_kernel<<<(n + 255) / 256, 256, 0, stream>>>(src, dst, n);
  };
  conv(t_qkv, wq_t, 1536 * 512);
  conv(i_qkv, wq_i, 1536 * 512);
  conv(t_ow, wo_t, 512 * 512);
  conv(i_ow, wo_i, 512 * 512);
  conv(t_f1w, w1_t, CFF * CD);
  conv(i_f1w, w1_i, CFF * CD);
  conv(t_f2w, w2_t, CD * CFF);
  conv(i_f2w, w2_i, CD * CFF);

  // 2. adaLN embeddings
  adaln_emb_kernel<<<CB * 6 * CD / 8, 256, 0, stream>>>(vec, i_aw, i_ab, e_img);
  adaln_emb_kernel<<<CB * 6 * CD / 8, 256, 0, stream>>>(vec, t_aw, t_ab, e_txt);

  // 3. per-batch effective scale, spatial modulation
  eff_kernel<<<1, 32, 0, stream>>>(soltp, effb);
  mod_kernel<<<CB * CNH * CN / 256, 256, 0, stream>>>(solsp, modw, modb, modbf);

  // 4. pre-attention norms (shift=sm@0, scale=cm@D)
  norm_mod_kernel<<<CB * CN, 256, 0, stream>>>(img, e_img, i_anw, imgn, CN, 0, CD);
  norm_mod_kernel<<<CB * CL, 256, 0, stream>>>(txt, e_txt, t_anw, txtn, CL, 0, CD);

  // 5. QKV GEMMs
  gemm_bf16_kernel<<<dim3(1536 / 128, CB * CL / 128), 256, 0, stream>>>(
      txtn, wq_t, nullptr, tqkv, nullptr, nullptr, nullptr,
      CB * CL, 1536, CD, 0, CL);
  gemm_bf16_kernel<<<dim3(1536 / 128, CB * CN / 128), 256, 0, stream>>>(
      imgn, wq_i, nullptr, iqkv, nullptr, nullptr, nullptr,
      CB * CN, 1536, CD, 0, CN);

  // 6. assemble Q/K/V with rope + mod (V transposed)
  assemble_kernel<<<dim3(CT, CNH, CB), 128, 0, stream>>>(
      tqkv, iqkv, rope, modbf, Qb, Kb, Vtb);

  // 7. joint flash attention
  attn_kernel<<<dim3(CT / 16, CNH, CB), 32, 0, stream>>>(
      Qb, Kb, Vtb, effb, tattn, iattn);

  // 8. output projections + gated residual (gate = gm @ 2D)
  gemm_bf16_kernel<<<dim3(CD / 128, CB * CL / 128), 256, 0, stream>>>(
      tattn, wo_t, nullptr, txt1, nullptr, txt, e_txt + 2 * CD,
      CB * CL, CD, CD, 2, CL);
  gemm_bf16_kernel<<<dim3(CD / 128, CB * CN / 128), 256, 0, stream>>>(
      iattn, wo_i, nullptr, img1, nullptr, img, e_img + 2 * CD,
      CB * CN, CD, CD, 2, CN);

  // 9. pre-MLP norms (shift=sp@3D, scale=cp@4D)
  norm_mod_kernel<<<CB * CL, 256, 0, stream>>>(txt1, e_txt, t_n2w, txtin, CL, 3 * CD, 4 * CD);
  norm_mod_kernel<<<CB * CN, 256, 0, stream>>>(img1, e_img, i_n2w, imgin, CN, 3 * CD, 4 * CD);

  // 10. fc1 GEMMs (bias + gelu -> bf16)
  gemm_bf16_kernel<<<dim3(CFF / 128, CB * CL / 128), 256, 0, stream>>>(
      txtin, w1_t, t_f1b, nullptr, h_t, nullptr, nullptr,
      CB * CL, CFF, CD, 1, CL);
  gemm_bf16_kernel<<<dim3(CFF / 128, CB * CN / 128), 256, 0, stream>>>(
      imgin, w1_i, i_f1b, nullptr, h_i, nullptr, nullptr,
      CB * CN, CFF, CD, 1, CN);

  // 11. fc2 GEMMs (bias + gate(gp@5D) + residual) -> d_out
  float* out_txt = out;                              // [B, L, D]
  float* out_img = out + (size_t)CB * CL * CD;       // [B, N, D]
  gemm_bf16_kernel<<<dim3(CD / 128, CB * CL / 128), 256, 0, stream>>>(
      h_t, w2_t, t_f2b, out_txt, nullptr, txt1, e_txt + 5 * CD,
      CB * CL, CD, CFF, 2, CL);
  gemm_bf16_kernel<<<dim3(CD / 128, CB * CN / 128), 256, 0, stream>>>(
      h_i, w2_i, i_f2b, out_img, nullptr, img1, e_img + 5 * CD,
      CB * CN, CD, CFF, 2, CN);

  (void)in_sizes; (void)n_in; (void)out_size; (void)ws_size;
}
